// ScaleDotProductAttention_61727269978516
// MI455X (gfx1250) — matrix operand order repaired and checked by reading, not compiled, not run on hardware
//
#include <hip/hip_runtime.h>
#include <hip/hip_bf16.h>

typedef __attribute__((ext_vector_type(16))) _Float16 v16h;
typedef __attribute__((ext_vector_type(8)))  float    v8f;
typedef __attribute__((ext_vector_type(4)))  float    f4;
typedef __attribute__((ext_vector_type(4)))  int      i4;

#define BB 2
#define HH 16
#define LL 2048
#define DD 64
#define ROWS 32
#define PITCH 2052            // LL + 4 floats of padding (kills LDS bank conflicts)
#define NTHREADS 256          // 8 wave32

// Convert 16 floats (8 at p0, 8 at p1) to a v16h fragment, scaling on the way.
__device__ __forceinline__ v16h cvt_h16(const float* p0, const float* p1, float scale) {
  f4 a0 = *(const f4*)(p0);
  f4 a1 = *(const f4*)(p0 + 4);
  f4 b0 = *(const f4*)(p1);
  f4 b1 = *(const f4*)(p1 + 4);
  v16h f;
  f[0]  = (_Float16)(a0.x * scale); f[1]  = (_Float16)(a0.y * scale);
  f[2]  = (_Float16)(a0.z * scale); f[3]  = (_Float16)(a0.w * scale);
  f[4]  = (_Float16)(a1.x * scale); f[5]  = (_Float16)(a1.y * scale);
  f[6]  = (_Float16)(a1.z * scale); f[7]  = (_Float16)(a1.w * scale);
  f[8]  = (_Float16)(b0.x * scale); f[9]  = (_Float16)(b0.y * scale);
  f[10] = (_Float16)(b0.z * scale); f[11] = (_Float16)(b0.w * scale);
  f[12] = (_Float16)(b1.x * scale); f[13] = (_Float16)(b1.y * scale);
  f[14] = (_Float16)(b1.z * scale); f[15] = (_Float16)(b1.w * scale);
  return f;
}

// V [B,H,L,D] f32  ->  Vt [B,H,D,L] f16 (coalesced via LDS tile transpose)
__global__ void vt_prep_kernel(const float* __restrict__ V, _Float16* __restrict__ Vt) {
  __shared__ _Float16 tile[16][18];
  const int tx = threadIdx.x, ty = threadIdx.y;
  const int lb = blockIdx.x * 16;
  const int db = blockIdx.y * 16;
  const int bh = blockIdx.z;
  tile[ty][tx] = (_Float16)V[((size_t)bh * LL + (lb + ty)) * DD + db + tx];
  __syncthreads();
  Vt[((size_t)bh * DD + (db + ty)) * LL + lb + tx] = tile[tx][ty];
}

__launch_bounds__(NTHREADS)
__global__ void attn_kernel(const float* __restrict__ Q, const float* __restrict__ K,
                            const int* __restrict__ Mask, const _Float16* __restrict__ Vt,
                            float* __restrict__ Out) {
  extern __shared__ float sS[];          // ROWS * PITCH floats (dynamic LDS, ~264 KB)
  __shared__ float sRed[NTHREADS];
  __shared__ float sRowMax[ROWS];
  __shared__ float sInvDen[ROWS];

  const int tid  = threadIdx.x;
  const int wave = tid >> 5;
  const int lane = tid & 31;
  const int m    = lane & 15;
  const int hi   = lane >> 4;

  const int bh = blockIdx.x >> 6;        // 64 query blocks per (b,h)
  const int qb = blockIdx.x & 63;
  const int b  = bh >> 4;                // H == 16
  const int q0 = qb * ROWS;

  // ---------------- Phase 1: S = (Q/8) @ K^T  ->  LDS f32 ----------------
  // A fragment (16x32 f16): lane m holds row M=m; elems 0..15 = K(d) hi*16+0..15.
  const float* Qb = Q + ((size_t)bh * LL + q0) * DD;
  v16h qf[2][2];
  #pragma unroll
  for (int qt = 0; qt < 2; ++qt) {
    const float* qrow = Qb + (qt * 16 + m) * DD;
    #pragma unroll
    for (int dh = 0; dh < 2; ++dh) {
      const float* p = qrow + dh * 32 + hi * 16;
      qf[qt][dh] = cvt_h16(p, p + 8, 0.125f);   // fold 1/sqrt(64) into Q
    }
  }
  // B fragment (32x16 f16): lane holds column N=m; elems 0..15 = K(d) hi*16+0..15
  const float* Kb = K + (size_t)bh * LL * DD;
  for (int kt = wave; kt < LL / 16; kt += NTHREADS / 32) {
    const float* krow = Kb + (kt * 16 + m) * DD;
    v16h kf[2];
    #pragma unroll
    for (int dh = 0; dh < 2; ++dh) {
      const float* p = krow + dh * 32 + hi * 16;
      kf[dh] = cvt_h16(p, p + 8, 1.0f);
    }
    #pragma unroll
    for (int qt = 0; qt < 2; ++qt) {
      v8f c = {};
      c = __builtin_amdgcn_wmma_f32_16x16x32_f16(false, qf[qt][0], false, kf[0],
                                                 (short)0, c, false, false);
      c = __builtin_amdgcn_wmma_f32_16x16x32_f16(false, qf[qt][1], false, kf[1],
                                                 (short)0, c, false, false);
      // C/D: row = qt*16 + i + 8*hi, col = kt*16 + m
      float* dst = &sS[(qt * 16 + hi * 8) * PITCH + kt * 16 + m];
      #pragma unroll
      for (int i = 0; i < 8; ++i) dst[i * PITCH] = c[i];
    }
  }
  __syncthreads();

  // ---------------- Phase 2: apply mask (coalesced int4 reads) ----------------
  const int* Mb = Mask + ((size_t)b * LL + q0) * LL;   // mask[b,0,row,col]
  for (int r = 0; r < ROWS; ++r) {
    const int* mp = Mb + (size_t)r * LL;
    float* sp = &sS[r * PITCH];
    for (int c0 = tid * 4; c0 < LL; c0 += NTHREADS * 4) {
      i4 mv = *(const i4*)(mp + c0);
      f4 sv = *(f4*)(sp + c0);
      sv.x = mv.x ? sv.x : -10000.0f;
      sv.y = mv.y ? sv.y : -10000.0f;
      sv.z = mv.z ? sv.z : -10000.0f;
      sv.w = mv.w ? sv.w : -10000.0f;
      *(f4*)(sp + c0) = sv;
    }
  }
  __syncthreads();

  // ---------------- Phase 3: row max, exp (in place), row sum ----------------
  const int r   = tid >> 3;   // 8 threads per row
  const int sub = tid & 7;
  float* rowp = &sS[r * PITCH];
  float mx = -3.0e38f;
  for (int c0 = sub * 4; c0 < LL; c0 += 32) {
    f4 vv = *(const f4*)(rowp + c0);
    mx = fmaxf(mx, fmaxf(fmaxf(vv.x, vv.y), fmaxf(vv.z, vv.w)));
  }
  sRed[tid] = mx;
  __syncthreads();
  if (sub == 0) {
    float v0 = sRed[tid];
    #pragma unroll
    for (int j = 1; j < 8; ++j) v0 = fmaxf(v0, sRed[tid + j]);
    sRowMax[r] = v0;
  }
  __syncthreads();
  const float rm = sRowMax[r];
  float sum = 0.0f;
  for (int c0 = sub * 4; c0 < LL; c0 += 32) {
    f4 vv = *(const f4*)(rowp + c0);
    vv.x = __expf(vv.x - rm);
    vv.y = __expf(vv.y - rm);
    vv.z = __expf(vv.z - rm);
    vv.w = __expf(vv.w - rm);
    *(f4*)(rowp + c0) = vv;
    sum += vv.x + vv.y + vv.z + vv.w;
  }
  sRed[tid] = sum;
  __syncthreads();
  if (sub == 0) {
    float v0 = 0.0f;
    #pragma unroll
    for (int j = 0; j < 8; ++j) v0 += sRed[tid + j];
    sInvDen[r] = 1.0f / v0;
  }
  __syncthreads();

  // ---------------- Phase 4: write normalized attn (coalesced f4 stores) ------
  float* attnO = Out + (size_t)BB * HH * LL * DD + ((size_t)bh * LL + q0) * LL;
  for (int rr = 0; rr < ROWS; ++rr) {
    const float inv = sInvDen[rr];
    const float* sp = &sS[rr * PITCH];
    float* gp = attnO + (size_t)rr * LL;
    for (int c0 = tid * 4; c0 < LL; c0 += NTHREADS * 4) {
      f4 e = *(const f4*)(sp + c0);
      e.x *= inv; e.y *= inv; e.z *= inv; e.w *= inv;
      *(f4*)(gp + c0) = e;
    }
  }

  // ---------------- Phase 5: out = (E @ V) * invDen (f16 WMMA) ----------------
  // Each wave owns one 16x16 out tile: qt = wave>>2, dt = wave&3.
  const int qt = wave >> 2;
  const int dt = wave & 3;
  const _Float16* vrow = Vt + ((size_t)bh * DD + dt * 16 + m) * LL;  // B: col N=m
  const float*    erow = &sS[(qt * 16 + m) * PITCH];                 // A: row M=m
  v8f acc = {};
  for (int ks = 0; ks < LL / 32; ++ks) {
    const float* ep = erow + ks * 32 + hi * 16;
    v16h a = cvt_h16(ep, ep + 8, 1.0f);               // LDS f32 -> f16 A frag
    v16h bf = *(const v16h*)(vrow + ks * 32 + hi * 16); // contiguous 32B B frag
    acc = __builtin_amdgcn_wmma_f32_16x16x32_f16(false, a, false, bf,
                                                 (short)0, acc, false, false);
  }
  float* op = Out + ((size_t)bh * LL + q0 + qt * 16 + hi * 8) * DD + dt * 16 + m;
  #pragma unroll
  for (int i = 0; i < 8; ++i) {
    op[(size_t)i * DD] = acc[i] * sInvDen[qt * 16 + hi * 8 + i];
  }
}

extern "C" void kernel_launch(void* const* d_in, const int* in_sizes, int n_in,
                              void* d_out, int out_size, void* d_ws, size_t ws_size,
                              hipStream_t stream) {
  const float* q    = (const float*)d_in[0];
  const float* k    = (const float*)d_in[1];
  const float* v    = (const float*)d_in[2];
  const int*   mask = (const int*)d_in[3];
  float* out = (float*)d_out;
  _Float16* vt = (_Float16*)d_ws;   // 8 MB: [B,H,D,L] f16

  vt_prep_kernel<<<dim3(LL / 16, DD / 16, BB * HH), dim3(16, 16), 0, stream>>>(v, vt);

  const size_t lds_bytes = (size_t)ROWS * PITCH * sizeof(float);
  attn_kernel<<<dim3(BB * HH * (LL / ROWS)), dim3(NTHREADS), lds_bytes, stream>>>(
      q, k, mask, vt, out);
}
